// MambaBlock_37812892074509
// MI455X (gfx1250) — matrix operand order repaired and checked by reading, not compiled, not run on hardware
//
#include <hip/hip_runtime.h>
#include <hip/hip_bf16.h>
#include <math.h>

// Problem constants (from reference)
#define BATCH 512
#define HIDN  2048
#define DINN  4096
#define NSTN  16
#define DTRN  128
#define KWIN  4

typedef __attribute__((ext_vector_type(16))) _Float16 v16h;
typedef __attribute__((ext_vector_type(8)))  _Float16 v8h;
typedef __attribute__((ext_vector_type(4)))  _Float16 v4h;
typedef __attribute__((ext_vector_type(8)))  float    v8f;
typedef __attribute__((ext_vector_type(4)))  float    v4f;
typedef __attribute__((ext_vector_type(4)))  unsigned int v4u;
typedef __attribute__((ext_vector_type(8)))  int      v8i;
typedef __attribute__((ext_vector_type(4)))  int      v4i;

// Tensor Data Mover availability (this toolchain: 6-arg form,
// (uint32x4 g0, int32x8 g1, int32x4 g2, int32x4 g3, int32x8, i32 cpol))
#if defined(__has_builtin)
#if __has_builtin(__builtin_amdgcn_tensor_load_to_lds) && \
    __has_builtin(__builtin_amdgcn_s_wait_tensorcnt)
#define USE_TDM 1
#endif
#endif
#ifndef USE_TDM
#define USE_TDM 0
#endif

// ---------------------------------------------------------------------------
// WMMA GEMM:  C[m,n] = sum_k A[m,k] * B[n,k]   (A: MxK, B: NxK, C: MxN)
//
// EDGE=false (all dims tile-aligned, no guards):
//   block tile 128x256, BK=32, 8 wave32s as 2(M) x 4(N); each wave owns a
//   64x64 slab = 16 v_wmma_f32_16x16x32_f16 per K-step (16 WMMA : 16 ds_b128).
//   If A_HALF (A already f16), the A tile is fetched by the Tensor Data Mover
//   (tensor_load_to_lds, D# pad fields reproduce the 40-half LDS row pitch),
//   drained with s_wait_tensorcnt before the block barrier.
// EDGE=true: 128x128 guarded tile (used only for the N=160 GEMM).
// ---------------------------------------------------------------------------
template <bool A_HALF, bool EDGE>
__global__ __launch_bounds__(256) void gemm_tn_wmma(
    const void* __restrict__ Ap, const float* __restrict__ Bp,
    float* __restrict__ Cp, int M, int N, int K, int lda, int ldb, int ldc)
{
    constexpr int BM = 128;
    constexpr int BN = EDGE ? 128 : 256;
    constexpr int BK = 32;
    constexpr int LDSL = 40;                 // 32 halves + 8-half pad per row
    constexpr int NI = EDGE ? 2 : 4;         // 16x16 tiles per wave in N

    __shared__ __align__(16) _Float16 As[BM * LDSL];
    __shared__ __align__(16) _Float16 Bs[BN * LDSL];

    const int tid   = threadIdx.x;
    const int lane  = tid & 31;
    const int wave  = tid >> 5;
    const int lo    = lane & 15;
    const int hi    = lane >> 4;
    const int waveM = (wave & 1) * 64;                       // 2 waves down
    const int waveN = (wave >> 1) * (EDGE ? 32 : 64);        // 4 waves across
    const int tileM = blockIdx.y * BM;
    const int tileN = blockIdx.x * BN;

    v8f acc[4][NI] = {};

    for (int k0 = 0; k0 < K; k0 += BK) {
        // ---------------- A tile: BM x 32 halves ----------------
        if constexpr (A_HALF && !EDGE) {
#if USE_TDM
            if (tid < 32) {   // wave 0 drives the Tensor Data Mover
                const unsigned long long gaddr =
                    (unsigned long long)(const _Float16*)Ap +
                    ((unsigned long long)tileM * (unsigned)lda + (unsigned)k0) * 2ull;
                const unsigned int lds_off = (unsigned int)(size_t)(void*)As;
                v4u g0;
                g0[0] = 1u;                                   // count=1, user mode
                g0[1] = lds_off;                              // D#.lds_addr
                g0[2] = (unsigned int)(gaddr & 0xffffffffull);
                g0[3] = (unsigned int)((gaddr >> 32) & 0x01ffffffull) | (2u << 30); // type=2
                const unsigned long long str = (unsigned long long)(unsigned)lda;
                v8i g1;
                g1[0] = (int)((1u << 16)      // data_size = 2 bytes
                            | (1u << 20)      // pad_enable
                            | (3u << 22)      // pad_interval: 16 DWORDs (=32 halves)
                            | (1u << 25));    // pad_amount:   2 DWORDs (=4 halves)
                g1[1] = (int)(((unsigned)K & 0xffffu) << 16);             // tensor_dim0 lo
                g1[2] = (int)(((unsigned)K >> 16) | (((unsigned)M & 0xffffu) << 16));
                g1[3] = (int)(((unsigned)M >> 16) | (32u << 16));         // tile_dim0 = 32
                g1[4] = (int)(unsigned)BM;                                // tile_dim1 = 128
                g1[5] = (int)(unsigned)(str & 0xffffffffull);             // dim0 stride lo
                g1[6] = (int)(unsigned)(str >> 32);                       // dim0 stride hi
                g1[7] = 0;
                v4i gz4 = {0, 0, 0, 0};
                v8i gz8 = {0, 0, 0, 0, 0, 0, 0, 0};
                __builtin_amdgcn_tensor_load_to_lds(g0, g1, gz4, gz4, gz8, 0);
                __builtin_amdgcn_s_wait_tensorcnt(0);
            }
#else
            {   // manual f16 copy: 16 halves/thread
                const int r = tid >> 1, c = (tid & 1) * 16;
                const _Float16* src = (const _Float16*)Ap + (size_t)(tileM + r) * lda + k0 + c;
                _Float16* dst = &As[r * LDSL + c];
                *(v8h*)(dst)     = *(const v8h*)(src);
                *(v8h*)(dst + 8) = *(const v8h*)(src + 8);
            }
#endif
        } else {
            const int r = tid >> 1, c = (tid & 1) * 16;
            const int m = tileM + r;
            _Float16* dst = &As[r * LDSL + c];
            bool ok = true;
            if constexpr (EDGE) ok = (m < M);
            if (ok) {
                const float* src = (const float*)Ap + (size_t)m * lda + k0 + c;
#pragma unroll
                for (int i = 0; i < 16; i += 4) {
                    v4f x = *(const v4f*)(src + i);
                    dst[i + 0] = (_Float16)x[0];
                    dst[i + 1] = (_Float16)x[1];
                    dst[i + 2] = (_Float16)x[2];
                    dst[i + 3] = (_Float16)x[3];
                }
                if (k0 + BK < K) __builtin_prefetch(src + BK, 0, 1);
            } else {
                v8h z = {};
                *(v8h*)(dst) = z; *(v8h*)(dst + 8) = z;
            }
        }
        // ---------------- B tile: BN x 32 halves ----------------
        if constexpr (!EDGE) {
            // BN=256: one full 32-half row per thread, never OOB
            const int n = tileN + tid;
            const float* src = Bp + (size_t)n * ldb + k0;
            _Float16* dst = &Bs[tid * LDSL];
#pragma unroll
            for (int i = 0; i < 32; i += 4) {
                v4f x = *(const v4f*)(src + i);
                dst[i + 0] = (_Float16)x[0];
                dst[i + 1] = (_Float16)x[1];
                dst[i + 2] = (_Float16)x[2];
                dst[i + 3] = (_Float16)x[3];
            }
            if (k0 + BK < K) __builtin_prefetch(src + BK, 0, 1);
        } else {
            const int r = tid >> 1, c = (tid & 1) * 16;
            const int n = tileN + r;
            _Float16* dst = &Bs[r * LDSL + c];
            if (n < N) {
                const float* src = Bp + (size_t)n * ldb + k0 + c;
#pragma unroll
                for (int i = 0; i < 16; i += 4) {
                    v4f x = *(const v4f*)(src + i);
                    dst[i + 0] = (_Float16)x[0];
                    dst[i + 1] = (_Float16)x[1];
                    dst[i + 2] = (_Float16)x[2];
                    dst[i + 3] = (_Float16)x[3];
                }
            } else {
                v8h z = {};
                *(v8h*)(dst) = z; *(v8h*)(dst + 8) = z;
            }
        }
        __syncthreads();

        // B fragments: 32x16 f16; lane = col lo, half i -> K = hi*16 + i
        v16h bfrag[NI];
#pragma unroll
        for (int ni = 0; ni < NI; ++ni) {
            const _Float16* bp = &Bs[(waveN + ni * 16 + lo) * LDSL + hi * 16];
            union { v16h v; v8h h[2]; } u;
            u.h[0] = *(const v8h*)(bp);
            u.h[1] = *(const v8h*)(bp + 8);
            bfrag[ni] = u.v;
        }
        // A fragments + WMMAs: A 16x32 f16; lane = row lo,
        // half i -> K = hi*16 + i
#pragma unroll
        for (int mi = 0; mi < 4; ++mi) {
            const _Float16* ap = &As[(waveM + mi * 16 + lo) * LDSL + hi * 16];
            union { v16h v; v8h h[2]; } u;
            u.h[0] = *(const v8h*)(ap);
            u.h[1] = *(const v8h*)(ap + 8);
            v16h afrag = u.v;
#pragma unroll
            for (int ni = 0; ni < NI; ++ni) {
                acc[mi][ni] = __builtin_amdgcn_wmma_f32_16x16x32_f16(
                    false, afrag, false, bfrag[ni], (short)0, acc[mi][ni], false, false);
            }
        }
        __syncthreads();
    }

    // store: D vgpr r -> (m = base + hi*8 + r, n = base + lo)
#pragma unroll
    for (int mi = 0; mi < 4; ++mi) {
#pragma unroll
        for (int ni = 0; ni < NI; ++ni) {
            const int n = tileN + waveN + ni * 16 + lo;
            if (EDGE && n >= N) continue;
#pragma unroll
            for (int r = 0; r < 8; ++r) {
                const int m = tileM + waveM + mi * 16 + hi * 8 + r;
                if (EDGE && m >= M) continue;
                Cp[(size_t)m * ldc + n] = acc[mi][ni][r];
            }
        }
    }
}

// ---------------------------------------------------------------------------
// Pack conv window to f16 with reduction index r = c*K + k (makes the conv
// einsum a plain GEMM against row-major W_conv), and emit inputs_new.
// ---------------------------------------------------------------------------
__global__ __launch_bounds__(256) void pack_window(
    const float* __restrict__ conv_inputs, const float* __restrict__ xz,
    _Float16* __restrict__ A2, float* __restrict__ inputs_new)
{
    const int idx = blockIdx.x * blockDim.x + threadIdx.x;  // b*DIN + c
    if (idx >= BATCH * DINN) return;
    const int b = idx / DINN, c = idx % DINN;
    const float w0 = conv_inputs[(size_t)b * 3 * DINN + 0 * DINN + c];
    const float w1 = conv_inputs[(size_t)b * 3 * DINN + 1 * DINN + c];
    const float w2 = conv_inputs[(size_t)b * 3 * DINN + 2 * DINN + c];
    const float xp = xz[(size_t)b * 2 * DINN + c];           // x_part
    v4h pk = { (_Float16)w0, (_Float16)w1, (_Float16)w2, (_Float16)xp };
    *(v4h*)&A2[(size_t)b * (DINN * KWIN) + (size_t)c * KWIN] = pk;
    float* ip = inputs_new + (size_t)b * 3 * DINN;
    ip[0 * DINN + c] = w1;
    ip[1 * DINN + c] = w2;
    ip[2 * DINN + c] = xp;
}

// xs = silu(x_conv + b_conv), in place
__global__ __launch_bounds__(256) void bias_silu(
    float* __restrict__ xc, const float* __restrict__ b_conv)
{
    const int idx = blockIdx.x * blockDim.x + threadIdx.x;
    if (idx >= BATCH * DINN) return;
    const float v = xc[idx] + b_conv[idx % DINN];
    xc[idx] = v / (1.f + __expf(-v));
}

// ---------------------------------------------------------------------------
// Fused bandwidth kernel: softplus(delta) -> h_new recurrence -> y -> gate.
// One thread per (b,d): 64B contiguous h load + 64B h_new store.
// ---------------------------------------------------------------------------
__global__ __launch_bounds__(256) void state_update(
    const float* __restrict__ dpre, const float* __restrict__ b_dt,
    const float* __restrict__ xs,   const float* __restrict__ dbc,
    const float* __restrict__ A_log, const float* __restrict__ Dv,
    const float* __restrict__ h,    const float* __restrict__ xz,
    float* __restrict__ h_new, float* __restrict__ g)
{
    __shared__ float sB[NSTN], sC[NSTN];
    const int b = blockIdx.x / (DINN / 256);
    const int d = (blockIdx.x % (DINN / 256)) * 256 + threadIdx.x;
    if (threadIdx.x < 2 * NSTN) {
        const float v = dbc[(size_t)b * (DTRN + 2 * NSTN) + DTRN + threadIdx.x];
        if (threadIdx.x < NSTN) sB[threadIdx.x] = v; else sC[threadIdx.x - NSTN] = v;
    }
    __syncthreads();

    const size_t bd = (size_t)b * DINN + d;
    const float xr  = dpre[bd] + b_dt[d];
    const float dlt = (xr > 20.f) ? xr : log1pf(__expf(xr));   // softplus
    const float xsv = xs[bd];
    const float dx  = dlt * xsv;
    const float* hp = h + bd * NSTN;
    float* hq       = h_new + bd * NSTN;
    float y = 0.f;
#pragma unroll
    for (int n0 = 0; n0 < NSTN; n0 += 4) {
        v4f hv = *(const v4f*)(hp + n0);
        v4f al = *(const v4f*)(&A_log[(size_t)d * NSTN + n0]);
        v4f hn;
#pragma unroll
        for (int j = 0; j < 4; ++j) {
            const float a  = -__expf(al[j]);
            const float dA = __expf(dlt * a);
            const float v  = dA * hv[j] + dx * sB[n0 + j];
            hn[j] = v;
            y += v * sC[n0 + j];
        }
        *(v4f*)(hq + n0) = hn;
    }
    y += Dv[d] * xsv;
    const float zv = xz[(size_t)b * 2 * DINN + DINN + d];
    g[bd] = y * (zv / (1.f + __expf(-zv)));                    // y * silu(z)
}

// ---------------------------------------------------------------------------
extern "C" void kernel_launch(void* const* d_in, const int* in_sizes, int n_in,
                              void* d_out, int out_size, void* d_ws, size_t ws_size,
                              hipStream_t stream)
{
    const float* x         = (const float*)d_in[0];
    const float* h         = (const float*)d_in[1];
    const float* conv_in   = (const float*)d_in[2];
    const float* W_in      = (const float*)d_in[3];
    const float* W_conv    = (const float*)d_in[4];
    const float* b_conv    = (const float*)d_in[5];
    const float* W_x       = (const float*)d_in[6];
    const float* W_dt      = (const float*)d_in[7];
    const float* b_dt      = (const float*)d_in[8];
    const float* A_log     = (const float*)d_in[9];
    const float* Dv        = (const float*)d_in[10];
    const float* W_out     = (const float*)d_in[11];

    float* out        = (float*)d_out;                             // 512x2048
    float* h_new      = out + (size_t)BATCH * HIDN;                // 512x4096x16
    float* inputs_new = h_new + (size_t)BATCH * DINN * NSTN;       // 512x3x4096

    char* w = (char*)d_ws;
    float*    xz    = (float*)w;     w += (size_t)BATCH * 2 * DINN * 4;    // 16 MB
    _Float16* A2    = (_Float16*)w;  w += (size_t)BATCH * DINN * KWIN * 2; // 16 MB
    float*    xconv = (float*)w;     w += (size_t)BATCH * DINN * 4;        //  8 MB
    float*    dbc   = (float*)w;     w += (size_t)BATCH * (DTRN + 2 * NSTN) * 4;
    float*    dpre  = (float*)w;     w += (size_t)BATCH * DINN * 4;        //  8 MB
    float*    g     = (float*)w;                                           //  8 MB

    const dim3 blk(256);
    const int NDC = DTRN + 2 * NSTN;  // 160

    // 1) xz = x @ W_in^T              (512 x 8192, K=2048)
    gemm_tn_wmma<false, false><<<dim3(2 * DINN / 256, BATCH / 128), blk, 0, stream>>>(
        x, W_in, xz, BATCH, 2 * DINN, HIDN, HIDN, HIDN, 2 * DINN);
    // 2) pack window (f16) + inputs_new
    pack_window<<<(BATCH * DINN + 255) / 256, blk, 0, stream>>>(conv_in, xz, A2, inputs_new);
    // 3) x_conv = window @ W_conv_flat^T   (512 x 4096, K=16384) -- TDM feeds A
    gemm_tn_wmma<true, false><<<dim3(DINN / 256, BATCH / 128), blk, 0, stream>>>(
        A2, W_conv, xconv, BATCH, DINN, DINN * KWIN, DINN * KWIN, DINN * KWIN, DINN);
    // 4) xs = silu(x_conv + b_conv)
    bias_silu<<<(BATCH * DINN + 255) / 256, blk, 0, stream>>>(xconv, b_conv);
    // 5) dbc = xs @ W_x^T             (512 x 160, K=4096) -- guarded edge kernel
    gemm_tn_wmma<false, true><<<dim3((NDC + 127) / 128, BATCH / 128), blk, 0, stream>>>(
        xconv, W_x, dbc, BATCH, NDC, DINN, DINN, DINN, NDC);
    // 6) dpre = delta_r @ W_dt^T      (512 x 4096, K=128; A strided inside dbc)
    gemm_tn_wmma<false, false><<<dim3(DINN / 256, BATCH / 128), blk, 0, stream>>>(
        dbc, W_dt, dpre, BATCH, DINN, DTRN, NDC, DTRN, DINN);
    // 7) fused recurrence: h_new, g = y * silu(z)
    state_update<<<BATCH * (DINN / 256), blk, 0, stream>>>(
        dpre, b_dt, xconv, dbc, A_log, Dv, h, xz, h_new, g);
    // 8) out = g @ W_out^T            (512 x 2048, K=4096)
    gemm_tn_wmma<false, false><<<dim3(HIDN / 256, BATCH / 128), blk, 0, stream>>>(
        g, W_out, out, BATCH, HIDN, DINN, DINN, DINN, HIDN);
}
